// TransformerBlock_2078764172055
// MI455X (gfx1250) — matrix operand order repaired and checked by reading, not compiled, not run on hardware
//
#include <hip/hip_runtime.h>

// ---------------- problem constants ----------------
#define SHIFT   3
#define NWIN    2048           // B * (56/7)^2
#define ROWS    100352         // NWIN * 49
#define QKV_N   384
#define SCALE_Q 0.17677669529663687f  // 32^-0.5

typedef _Float16 half16v __attribute__((ext_vector_type(16)));
typedef _Float16 half4v  __attribute__((ext_vector_type(4)));
typedef _Float16 half2v  __attribute__((ext_vector_type(2)));
typedef float    float8v __attribute__((ext_vector_type(8)));

union H16 { half16v v; half2v p[8]; _Float16 e[16]; };
union F8  { float8v v; float e[8]; };

// ---------------- WMMA helper (CDNA5 wave32, 16x16x32 f16) ----------------
__device__ inline float8v wmma32(half16v a, half16v b, float8v c) {
  return __builtin_amdgcn_wmma_f32_16x16x32_f16(false, a, false, b, (short)0, c,
                                                false, false);
}

// A tile: row-major A[M][lda], tile at (m0, k0). contiguous K order (k = 16h + s), not the ISA table order:
// lane 0..15 -> M=lane, K pairs {0,2,...,14}; lanes 16..31 get K +16.
// Branchless padding: clamp row address, select-zero the value.
__device__ inline half16v load_a_tile(const _Float16* A, int lda, int m0, int k0,
                                      int Mbound) {
  int lane = threadIdx.x & 31;
  int m = m0 + (lane & 15);
  bool ok = m < Mbound;
  int mc = ok ? m : (Mbound - 1);
  int kadd = (lane >= 16) ? 16 : 0;
  const _Float16* row = A + (long)mc * lda;
  const half2v z = {};
  H16 a;
#pragma unroll
  for (int j = 0; j < 8; ++j) {
    int k = k0 + kadd + 2 * j;
    half2v t = *(const half2v*)(row + k);
    a.p[j] = ok ? t : z;
  }
  return a.v;
}

// B tile from transposed weights BT[N][ldk] (always in-bounds).
__device__ inline half16v load_bt_tile(const _Float16* BT, int ldk, int n0, int k0) {
  int lane = threadIdx.x & 31;
  int n = n0 + (lane & 15);
  int kb = k0 + ((lane >= 16) ? 16 : 0);
  const _Float16* col = BT + (long)n * ldk + kb;
  H16 b;
#pragma unroll
  for (int j = 0; j < 8; ++j) b.p[j] = *(const half2v*)(col + 2 * j);
  return b.v;
}

__device__ inline float wave_sum(float s) {
#pragma unroll
  for (int o = 16; o > 0; o >>= 1) s += __shfl_xor(s, o, 32);
  return s;
}

__device__ inline int reg3(int g) { return (g < 49) ? 0 : ((g < 53) ? 1 : 2); }

// ---------------- weight convert+transpose: wT[n*K+k] = (f16)w[k*N+n] -------
__global__ __launch_bounds__(256) void wcvt_kernel(const float* __restrict__ w,
                                                   _Float16* __restrict__ wT,
                                                   int K, int N) {
  int idx = blockIdx.x * 256 + threadIdx.x;
  if (idx >= K * N) return;
  int n = idx / K, k = idx % K;
  wT[idx] = (_Float16)w[(long)k * N + n];
}

// ---------------- LN1 + cyclic shift + window partition -> f16 --------------
__global__ __launch_bounds__(256) void ln1_kernel(const float* __restrict__ x,
                                                  const float* __restrict__ g,
                                                  const float* __restrict__ b,
                                                  _Float16* __restrict__ win) {
  int row = blockIdx.x * 8 + (threadIdx.x >> 5);     // 0..100351 windowed row
  int lane = threadIdx.x & 31;
  int widx = row / 49, p = row % 49;
  int bi = widx >> 6, wi = widx & 63;
  int hs = (wi >> 3) * 7 + p / 7;
  int wsc = (wi & 7) * 7 + p % 7;
  int sh = hs + SHIFT; if (sh >= 56) sh -= 56;       // roll(-SHIFT) source
  int sw = wsc + SHIFT; if (sw >= 56) sw -= 56;
  const float* src = x + ((long)bi * 3136 + sh * 56 + sw) * 128;
  float4 v = ((const float4*)src)[lane];
  float mu = wave_sum(v.x + v.y + v.z + v.w) * (1.0f / 128.0f);
  float d0 = v.x - mu, d1 = v.y - mu, d2 = v.z - mu, d3 = v.w - mu;
  float var = wave_sum(d0 * d0 + d1 * d1 + d2 * d2 + d3 * d3) * (1.0f / 128.0f);
  float rs = rsqrtf(var + 1e-5f);
  int c = lane * 4;
  half4v o;
  o.x = (_Float16)(d0 * rs * g[c + 0] + b[c + 0]);
  o.y = (_Float16)(d1 * rs * g[c + 1] + b[c + 1]);
  o.z = (_Float16)(d2 * rs * g[c + 2] + b[c + 2]);
  o.w = (_Float16)(d3 * rs * g[c + 3] + b[c + 3]);
  *(half4v*)(win + (long)row * 128 + c) = o;
}

// ---------------- LN2 (no permutation) --------------------------------------
__global__ __launch_bounds__(256) void ln2_kernel(const float* __restrict__ x1,
                                                  const float* __restrict__ g,
                                                  const float* __restrict__ b,
                                                  _Float16* __restrict__ h2) {
  int row = blockIdx.x * 8 + (threadIdx.x >> 5);
  int lane = threadIdx.x & 31;
  const float* src = x1 + (long)row * 128;
  float4 v = ((const float4*)src)[lane];
  float mu = wave_sum(v.x + v.y + v.z + v.w) * (1.0f / 128.0f);
  float d0 = v.x - mu, d1 = v.y - mu, d2 = v.z - mu, d3 = v.w - mu;
  float var = wave_sum(d0 * d0 + d1 * d1 + d2 * d2 + d3 * d3) * (1.0f / 128.0f);
  float rs = rsqrtf(var + 1e-5f);
  int c = lane * 4;
  half4v o;
  o.x = (_Float16)(d0 * rs * g[c + 0] + b[c + 0]);
  o.y = (_Float16)(d1 * rs * g[c + 1] + b[c + 1]);
  o.z = (_Float16)(d2 * rs * g[c + 2] + b[c + 2]);
  o.w = (_Float16)(d3 * rs * g[c + 3] + b[c + 3]);
  *(half4v*)(h2 + (long)row * 128 + c) = o;
}

// ---------------- QKV GEMM: 16x64 strip per wave, [.,128]x[128,384]+bias ----
__global__ __launch_bounds__(256) void qkv_gemm_kernel(
    const _Float16* __restrict__ A, const _Float16* __restrict__ BT,
    const float* __restrict__ bias, _Float16* __restrict__ out) {
  int wid = blockIdx.x * 8 + (threadIdx.x >> 5);
  int tm = wid / 6, n0 = (wid % 6) * 64;            // 6272 x 6 strips
  F8 acc[4] = {};
  for (int k0 = 0; k0 < 128; k0 += 32) {
    half16v a = load_a_tile(A, 128, tm * 16, k0, 1 << 30);
#pragma unroll
    for (int t = 0; t < 4; ++t)
      acc[t].v = wmma32(a, load_bt_tile(BT, 128, n0 + t * 16, k0), acc[t].v);
  }
  int lane = threadIdx.x & 31;
  int m0 = tm * 16 + ((lane >= 16) ? 8 : 0);
#pragma unroll
  for (int t = 0; t < 4; ++t) {
    int n = n0 + t * 16 + (lane & 15);
    float bn = bias[n];
#pragma unroll
    for (int j = 0; j < 8; ++j)
      out[(long)(m0 + j) * QKV_N + n] = (_Float16)(acc[t].e[j] + bn);
  }
}

// ---------------- attention: 1 wave per (window, head) -----------------------
__global__ __launch_bounds__(64) void attn_kernel(const _Float16* __restrict__ qkv,
                                                  const float* __restrict__ relt,
                                                  _Float16* __restrict__ attn_out) {
  __shared__ float S[2][64][65];
  __shared__ _Float16 P[2][64][64];
  int wslot = threadIdx.x >> 5;
  int lane = threadIdx.x & 31;
  int pair = blockIdx.x * 2 + wslot;                // 0..8191
  int b_ = pair >> 2, h = pair & 3;
  const long THIRD = 12845056;                      // ROWS*384/3 elements
  long base = (long)b_ * 6272 + h * 1568;           // raw-reshape contiguity
  const _Float16* q = qkv + base;
  const _Float16* kmat = qkv + THIRD + base;
  const _Float16* vmat = qkv + 2 * THIRD + base;
  int wi = b_ & 63, wh = wi >> 3, ww = wi & 7;

  // ---- S = (q * SCALE) @ k^T + bias + mask ----
  for (int tm = 0; tm < 4; ++tm) {
    half16v a = load_a_tile(q, 32, tm * 16, 0, 49);
#pragma unroll
    for (int tn = 0; tn < 4; ++tn) {
      H16 bt;
      {
        int jcol = tn * 16 + (lane & 15);
        bool ok = jcol < 49;
        int jc = ok ? jcol : 48;                    // clamp address, select value
        int db = (lane >= 16) ? 16 : 0;
        const _Float16* krow = kmat + jc * 32 + db;
        const half2v z = {};
#pragma unroll
        for (int t = 0; t < 8; ++t) {
          half2v v = *(const half2v*)(krow + 2 * t);
          bt.p[t] = ok ? v : z;
        }
      }
      F8 acc{};
      acc.v = wmma32(a, bt.v, acc.v);
      int n = tn * 16 + (lane & 15);
      int m0 = tm * 16 + ((lane >= 16) ? 8 : 0);
#pragma unroll
      for (int j = 0; j < 8; ++j) {
        int m = m0 + j;
        float val = -1e30f;
        if (m < 49 && n < 49) {
          int mh = m / 7, mw = m % 7, nh = n / 7, nw = n % 7;
          int rel = (mh - nh + 6) * 13 + (mw - nw + 6);
          float bias = relt[rel * 4 + h];
          float msk = (reg3(wh * 7 + mh) == reg3(wh * 7 + nh) &&
                       reg3(ww * 7 + mw) == reg3(ww * 7 + nw)) ? 0.0f : -100.0f;
          val = acc.e[j] * SCALE_Q + bias + msk;
        }
        S[wslot][m][n] = val;                       // wave-in-order LDS, no barrier
      }
    }
  }

  // ---- softmax over 49 cols, zero-pad P to 64x64 f16 ----
  for (int r = lane; r < 64; r += 32) {
    if (r < 49) {
      float mx = -1e30f;
      for (int c = 0; c < 49; ++c) mx = fmaxf(mx, S[wslot][r][c]);
      float sum = 0.0f;
      for (int c = 0; c < 49; ++c) {
        float e = __expf(S[wslot][r][c] - mx);
        S[wslot][r][c] = e;
        sum += e;
      }
      float inv = 1.0f / sum;
      for (int c = 0; c < 64; ++c)
        P[wslot][r][c] = (c < 49) ? (_Float16)(S[wslot][r][c] * inv) : (_Float16)0;
    } else {
      for (int c = 0; c < 64; ++c) P[wslot][r][c] = (_Float16)0;
    }
  }

  // ---- O = P @ v  (K padded to 64) ----
  const _Float16* Pp = &P[wslot][0][0];
#pragma unroll
  for (int tm = 0; tm < 4; ++tm) {
#pragma unroll
    for (int tn = 0; tn < 2; ++tn) {
      F8 acc{};
#pragma unroll
      for (int ks = 0; ks < 2; ++ks) {
        half16v a = load_a_tile(Pp, 64, tm * 16, ks * 32, 64);
        H16 bt;
        {
          int d = tn * 16 + (lane & 15);
          int kb = ks * 32 + ((lane >= 16) ? 16 : 0);
#pragma unroll
          for (int t = 0; t < 8; ++t) {
            int jp = kb + 2 * t;
            int j0 = (jp < 48) ? jp : 48;           // clamped addresses
            int j1 = (jp + 1 < 48) ? (jp + 1) : 48;
            _Float16 lo = vmat[j0 * 32 + d];
            _Float16 hi = vmat[j1 * 32 + d];
            bt.e[2 * t]     = (jp < 49)     ? lo : (_Float16)0;
            bt.e[2 * t + 1] = (jp + 1 < 49) ? hi : (_Float16)0;
          }
        }
        acc.v = wmma32(a, bt.v, acc.v);
      }
      int d = tn * 16 + (lane & 15);
      int m0 = tm * 16 + ((lane >= 16) ? 8 : 0);
#pragma unroll
      for (int j = 0; j < 8; ++j) {
        int m = m0 + j;
        if (m < 49)                                  // heads concat on channel dim
          attn_out[((long)b_ * 49 + m) * 128 + h * 32 + d] = (_Float16)acc.e[j];
      }
    }
  }
}

// ---------------- proj GEMM + window reverse + roll + residual -> fp32 x1 ----
__global__ __launch_bounds__(256) void proj_gemm_kernel(
    const _Float16* __restrict__ A, const _Float16* __restrict__ BT,
    const float* __restrict__ bias, const float* __restrict__ xin,
    float* __restrict__ x1) {
  int wid = blockIdx.x * 8 + (threadIdx.x >> 5);
  int tm = wid >> 1, n0 = (wid & 1) * 64;           // 6272 x 2 strips
  F8 acc[4] = {};
  for (int k0 = 0; k0 < 128; k0 += 32) {
    half16v a = load_a_tile(A, 128, tm * 16, k0, 1 << 30);
#pragma unroll
    for (int t = 0; t < 4; ++t)
      acc[t].v = wmma32(a, load_bt_tile(BT, 128, n0 + t * 16, k0), acc[t].v);
  }
  int lane = threadIdx.x & 31;
  int m0 = tm * 16 + ((lane >= 16) ? 8 : 0);
#pragma unroll
  for (int t = 0; t < 4; ++t) {
    int n = n0 + t * 16 + (lane & 15);
    float bn = bias[n];
#pragma unroll
    for (int j = 0; j < 8; ++j) {
      int m = m0 + j;
      int widx = m / 49, p = m % 49;
      int bi = widx >> 6, wi = widx & 63;
      int hs = (wi >> 3) * 7 + p / 7, wsc = (wi & 7) * 7 + p % 7;
      int dh = hs + SHIFT; if (dh >= 56) dh -= 56;  // roll(+SHIFT) destination
      int dw = wsc + SHIFT; if (dw >= 56) dw -= 56;
      long dst = ((long)bi * 3136 + dh * 56 + dw) * 128 + n;
      x1[dst] = acc[t].e[j] + bn + xin[dst];
    }
  }
}

// ---------------- fc1 GEMM + exact GELU -> f16 -------------------------------
__global__ __launch_bounds__(256) void fc1_gemm_kernel(
    const _Float16* __restrict__ A, const _Float16* __restrict__ BT,
    const float* __restrict__ bias, _Float16* __restrict__ out) {
  int wid = blockIdx.x * 8 + (threadIdx.x >> 5);
  int tm = wid >> 3, n0 = (wid & 7) * 64;           // 6272 x 8 strips (N=512)
  F8 acc[4] = {};
  for (int k0 = 0; k0 < 128; k0 += 32) {
    half16v a = load_a_tile(A, 128, tm * 16, k0, 1 << 30);
#pragma unroll
    for (int t = 0; t < 4; ++t)
      acc[t].v = wmma32(a, load_bt_tile(BT, 128, n0 + t * 16, k0), acc[t].v);
  }
  int lane = threadIdx.x & 31;
  int m0 = tm * 16 + ((lane >= 16) ? 8 : 0);
#pragma unroll
  for (int t = 0; t < 4; ++t) {
    int n = n0 + t * 16 + (lane & 15);
    float bn = bias[n];
#pragma unroll
    for (int j = 0; j < 8; ++j) {
      float v = acc[t].e[j] + bn;
      float ge = 0.5f * v * (1.0f + erff(v * 0.70710678118654752f));
      out[(long)(m0 + j) * 512 + n] = (_Float16)ge;
    }
  }
}

// ---------------- fc2 GEMM + bias + residual -> fp32 d_out -------------------
__global__ __launch_bounds__(256) void fc2_gemm_kernel(
    const _Float16* __restrict__ A, const _Float16* __restrict__ BT,
    const float* __restrict__ bias, const float* __restrict__ x1,
    float* __restrict__ out) {
  int wid = blockIdx.x * 8 + (threadIdx.x >> 5);
  int tm = wid >> 1, n0 = (wid & 1) * 64;           // 6272 x 2 strips, K=512
  F8 acc[4] = {};
  for (int k0 = 0; k0 < 512; k0 += 32) {
    half16v a = load_a_tile(A, 512, tm * 16, k0, 1 << 30);
#pragma unroll
    for (int t = 0; t < 4; ++t)
      acc[t].v = wmma32(a, load_bt_tile(BT, 512, n0 + t * 16, k0), acc[t].v);
  }
  int lane = threadIdx.x & 31;
  int m0 = tm * 16 + ((lane >= 16) ? 8 : 0);
#pragma unroll
  for (int t = 0; t < 4; ++t) {
    int n = n0 + t * 16 + (lane & 15);
    float bn = bias[n];
#pragma unroll
    for (int j = 0; j < 8; ++j) {
      long idx = (long)(m0 + j) * 128 + n;
      out[idx] = acc[t].e[j] + bn + x1[idx];
    }
  }
}

// ---------------- launcher ---------------------------------------------------
extern "C" void kernel_launch(void* const* d_in, const int* in_sizes, int n_in,
                              void* d_out, int out_size, void* d_ws, size_t ws_size,
                              hipStream_t stream) {
  (void)in_sizes; (void)n_in; (void)out_size; (void)ws_size;
  const float* x      = (const float*)d_in[0];
  const float* qkv_w  = (const float*)d_in[1];
  const float* qkv_b  = (const float*)d_in[2];
  const float* proj_w = (const float*)d_in[3];
  const float* proj_b = (const float*)d_in[4];
  const float* relt   = (const float*)d_in[5];
  const float* ln1_g  = (const float*)d_in[6];
  const float* ln1_b  = (const float*)d_in[7];
  const float* ln2_g  = (const float*)d_in[8];
  const float* ln2_b  = (const float*)d_in[9];
  const float* fc1_w  = (const float*)d_in[10];
  const float* fc1_b  = (const float*)d_in[11];
  const float* fc2_w  = (const float*)d_in[12];
  const float* fc2_b  = (const float*)d_in[13];
  float* out = (float*)d_out;

  char* ws = (char*)d_ws;
  _Float16* wqkvT    = (_Float16*)(ws + 0);          //  98304 B
  _Float16* wprojT   = (_Float16*)(ws + 98304);      //  32768 B
  _Float16* wfc1T    = (_Float16*)(ws + 131072);     // 131072 B
  _Float16* wfc2T    = (_Float16*)(ws + 262144);     // 131072 B
  _Float16* win      = (_Float16*)(ws + 393216);     // 25.69 MB
  _Float16* qkv      = (_Float16*)(ws + 26083328);   // 77.07 MB
  _Float16* attn_out = (_Float16*)(ws + 103153664);  // 25.69 MB
  float*    x1       = (float*)(ws + 128843776);     // 51.38 MB (end 180224000)
  _Float16* h2 = win;                                // reuse (win dead after qkv)
  _Float16* hg = qkv;                                // reuse qkv+attn (102.76 MB)

  wcvt_kernel<<<192, 256, 0, stream>>>(qkv_w, wqkvT, 128, 384);
  wcvt_kernel<<<64,  256, 0, stream>>>(proj_w, wprojT, 128, 128);
  wcvt_kernel<<<256, 256, 0, stream>>>(fc1_w, wfc1T, 128, 512);
  wcvt_kernel<<<256, 256, 0, stream>>>(fc2_w, wfc2T, 512, 128);

  ln1_kernel<<<12544, 256, 0, stream>>>(x, ln1_g, ln1_b, win);
  qkv_gemm_kernel<<<4704, 256, 0, stream>>>(win, wqkvT, qkv_b, qkv);
  attn_kernel<<<4096, 64, 0, stream>>>(qkv, relt, attn_out);
  proj_gemm_kernel<<<1568, 256, 0, stream>>>(attn_out, wprojT, proj_b, x, x1);
  ln2_kernel<<<12544, 256, 0, stream>>>(x1, ln2_g, ln2_b, h2);
  fc1_gemm_kernel<<<6272, 256, 0, stream>>>(h2, wfc1T, fc1_b, hg);
  fc2_gemm_kernel<<<1568, 256, 0, stream>>>(hg, wfc2T, fc2_b, x1, out);
}
